// GemmaAttention_39608188404310
// MI455X (gfx1250) — compile-verified, matrix operand order checked by reading, not run on hardware
//
#include <hip/hip_runtime.h>
#include <hip/hip_bf16.h>
#include <stdint.h>

// ---------------- problem constants ----------------
#define S_LEN   2048
#define BATCH   2
#define HEADS   8
#define DHEAD   256
#define HID     2048
#define QK_SCALE 0.0625f   // 256^-0.5

typedef __bf16 bf16_t;
typedef __attribute__((ext_vector_type(16))) bf16_t v16bf;
typedef __attribute__((ext_vector_type(8)))  float  v8f;
typedef __attribute__((ext_vector_type(4)))  int    v4i;

union FragBF { v16bf v; uint4 u[2]; unsigned short s[16]; };

// ---------- feature probes (compile-safe fallbacks) ----------
#if defined(__has_builtin)
#if __has_builtin(__builtin_amdgcn_global_load_async_to_lds_b128)
#define ASYNC_COPY_OK 1
#endif
#endif
#ifndef ASYNC_COPY_OK
#define ASYNC_COPY_OK 0
#endif

#if defined(__has_builtin)
#if __has_builtin(__builtin_amdgcn_s_wait_asynccnt)
#define WAIT_ASYNC(n) __builtin_amdgcn_s_wait_asynccnt(n)
#endif
#endif
#ifndef WAIT_ASYNC
#if ASYNC_COPY_OK
#define WAIT_ASYNC(n) asm volatile("s_wait_asynccnt %0" ::"n"(n) : "memory")
#else
#define WAIT_ASYNC(n)
#endif
#endif

#if defined(__has_builtin)
#if __has_builtin(__builtin_amdgcn_cvt_pk_bf16_f32)
#define PK_CVT_OK 1
#endif
#endif
#ifndef PK_CVT_OK
#define PK_CVT_OK 0
#endif

__device__ __forceinline__ unsigned short f2bf(float f) {
  unsigned int u = __builtin_bit_cast(unsigned int, f);
  u += 0x7FFFu + ((u >> 16) & 1u);            // round to nearest even
  return (unsigned short)(u >> 16);
}

__device__ __forceinline__ unsigned int pack2bf(float a, float b) {
#if PK_CVT_OK
  auto t = __builtin_amdgcn_cvt_pk_bf16_f32(a, b);   // lo = a, hi = b
  unsigned int u;
  __builtin_memcpy(&u, &t, 4);
  return u;
#else
  return (unsigned int)f2bf(a) | ((unsigned int)f2bf(b) << 16);
#endif
}

// Copy 16 bytes global -> LDS (16B-aligned both sides).  Async DMA (ASYNCcnt)
// when available.  Offset/cpol are required immediates, so byte offsets are
// folded into the pointers and the immediates stay 0.
__device__ __forceinline__ void cp_b128(void* lds, const void* g) {
#if ASYNC_COPY_OK
  __builtin_amdgcn_global_load_async_to_lds_b128(
      (__attribute__((address_space(1))) v4i*)(void*)g,
      (__attribute__((address_space(3))) v4i*)lds, 0, 0);
#else
  *(uint4*)lds = *(const uint4*)g;
#endif
}

// WMMA 16-bit A/B fragment from an LDS row laid out [row][k] (k contiguous):
// lanes 0-15 hold k in {0..7,16..23}, lanes 16-31 base 8 (per ISA layout).
__device__ __forceinline__ v16bf load_frag(const unsigned short* rowp, int koff) {
  FragBF f;
  f.u[0] = *reinterpret_cast<const uint4*>(rowp + koff);
  f.u[1] = *reinterpret_cast<const uint4*>(rowp + koff + 16);
  return f.v;
}

__device__ __forceinline__ v8f v8f_zero() {
  v8f z = {0.f, 0.f, 0.f, 0.f, 0.f, 0.f, 0.f, 0.f};
  return z;
}

// ---------------- fp32 -> bf16 conversion pass (packed cvt) ----------------
__global__ void cvt_bf16_kernel(unsigned short* __restrict__ dst,
                                const float* __restrict__ src,
                                float scale, int n8) {
  int i = blockIdx.x * blockDim.x + threadIdx.x;
  if (i >= n8) return;
  const float4 f0 = *(const float4*)(src + (size_t)i * 8);
  const float4 f1 = *(const float4*)(src + (size_t)i * 8 + 4);
  uint4 o;
  o.x = pack2bf(f0.x * scale, f0.y * scale);
  o.y = pack2bf(f0.z * scale, f0.w * scale);
  o.z = pack2bf(f1.x * scale, f1.y * scale);
  o.w = pack2bf(f1.z * scale, f1.w * scale);
  *(uint4*)(dst + (size_t)i * 8) = o;
}

// ---------------- GEMM: C[M,N] = A[M,K] * W[N,K]^T  (bf16 in, f32 acc/out) --
// 128x128 tile / 256 threads (8 waves, each 32x64), TK=32, double-buffered LDS
// filled by async global->LDS b128 copies overlapped with WMMA.
#define TM 128
#define TN 128
#define TK 32
#define LDT 48   // padded row (elems): 96B stride, 16B multiple

__global__ __launch_bounds__(256)
void gemm_bf16_wmma(float* __restrict__ C, const unsigned short* __restrict__ A,
                    const unsigned short* __restrict__ W, int M, int N, int K) {
  __shared__ unsigned short As[2][TM][LDT];
  __shared__ unsigned short Bs[2][TN][LDT];

  const int tid   = threadIdx.x;
  const int lane  = tid & 31;
  const int wid   = tid >> 5;
  const int waveM = wid & 3;
  const int waveN = wid >> 2;
  const int bm    = blockIdx.y * TM;
  const int bn    = blockIdx.x * TN;

  const int lcol   = lane & 15;
  const int koff   = (lane < 16) ? 0 : 8;
  const int lhalf8 = (lane >> 4) * 8;

  const int lrow = tid >> 1;         // 0..127
  const int lkp  = (tid & 1) * 16;   // 0 or 16  (32B per thread per matrix)

  v8f acc[2][4];
#pragma unroll
  for (int i = 0; i < 2; ++i)
#pragma unroll
    for (int j = 0; j < 4; ++j) acc[i][j] = v8f_zero();

  auto issue_tile = [&](int k0, int buf) {
    const unsigned short* ga = A + (size_t)(bm + lrow) * K + k0 + lkp;
    const unsigned short* gb = W + (size_t)(bn + lrow) * K + k0 + lkp;
    cp_b128(&As[buf][lrow][lkp],     ga);
    cp_b128(&As[buf][lrow][lkp + 8], ga + 8);
    cp_b128(&Bs[buf][lrow][lkp],     gb);
    cp_b128(&Bs[buf][lrow][lkp + 8], gb + 8);
  };

  const int nk = K / TK;
  issue_tile(0, 0);
  for (int kt = 0; kt < nk; ++kt) {
    const int buf = kt & 1;
    if (kt + 1 < nk) {
      issue_tile((kt + 1) * TK, buf ^ 1);
      WAIT_ASYNC(4);         // tile kt complete (in-order), tile kt+1 in flight
    } else {
      WAIT_ASYNC(0);
    }
    __syncthreads();          // tile kt visible from all waves

    v16bf af[2], bfg[4];
#pragma unroll
    for (int wm = 0; wm < 2; ++wm)
      af[wm] = load_frag(As[buf][waveM * 32 + wm * 16 + lcol], koff);
#pragma unroll
    for (int wn = 0; wn < 4; ++wn)
      bfg[wn] = load_frag(Bs[buf][waveN * 64 + wn * 16 + lcol], koff);
#pragma unroll
    for (int wm = 0; wm < 2; ++wm)
#pragma unroll
      for (int wn = 0; wn < 4; ++wn)
        acc[wm][wn] = __builtin_amdgcn_wmma_f32_16x16x32_bf16(
            false, af[wm], false, bfg[wn], (short)0, acc[wm][wn], false, false);

    __syncthreads();          // release buf for refill at kt+1's issue
  }

#pragma unroll
  for (int wm = 0; wm < 2; ++wm)
#pragma unroll
    for (int wn = 0; wn < 4; ++wn)
#pragma unroll
      for (int r = 0; r < 8; ++r) {
        int m = bm + waveM * 32 + wm * 16 + r + lhalf8;
        int n = bn + waveN * 64 + wn * 16 + lcol;
        C[(size_t)m * N + n] = acc[wm][wn][r];
      }
}

// ---------------- RoPE (in place, fp32), x shaped [B*S, heads, 256] ---------
__global__ void rope_kernel(float* __restrict__ x, const float* __restrict__ fcos,
                            const float* __restrict__ fsin, int heads, int total) {
  int idx = blockIdx.x * blockDim.x + threadIdx.x;
  if (idx >= total) return;
  int per_row = heads * 128;
  int row = idx / per_row;           // b*S + s
  int rem = idx - row * per_row;
  int h = rem >> 7;
  int i = rem & 127;
  int s = row & (S_LEN - 1);
  float c  = fcos[s * 128 + i];
  float sn = fsin[s * 128 + i];
  float* p = x + (size_t)row * heads * 256 + h * 256 + i;
  float x1 = p[0], x2 = p[128];
  p[0]   = x1 * c - x2 * sn;
  p[128] = x1 * sn + x2 * c;
}

// ---------------- Flash attention: 4 waves/block, 64 q rows/block -----------
// bf16 inputs (Q pre-scaled); K tile async-DMA'd to LDS, V transposed while
// the K DMA is in flight; bf16 output for the wo GEMM.
#define KLDP 264   // 528B row stride (16B multiple, bank-spread)
#define VLDP 48
#define PLDP 48

__global__ __launch_bounds__(128)
void attn_kernel(unsigned short* __restrict__ out, const unsigned short* __restrict__ q,
                 const unsigned short* __restrict__ kc, const unsigned short* __restrict__ vc) {
  __shared__ unsigned short Klds[32][KLDP];       // [t][d] : B-frag for QK^T
  __shared__ unsigned short Vlds[DHEAD][VLDP];    // [d][t] : B-frag for PV
  __shared__ unsigned short Plds[4][16][PLDP];    // per-wave P staging

  const int tid  = threadIdx.x;
  const int lane = tid & 31;
  const int wv_  = tid >> 5;
  const int qb   = blockIdx.x;
  const int h    = blockIdx.y;
  const int b    = blockIdx.z;

  const int lcol   = lane & 15;
  const int koff   = (lane < 16) ? 0 : 8;
  const int lhalf8 = (lane >> 4) * 8;
  const int m0     = qb * 64 + wv_ * 16;

  // ---- Q fragments: direct b128 loads of pre-scaled bf16 ----
  v16bf qf[8];
  {
    const unsigned short* qrow =
        q + ((size_t)(b * S_LEN + m0 + lcol)) * (HEADS * DHEAD) + h * DHEAD;
#pragma unroll
    for (int ds = 0; ds < 8; ++ds) {
      FragBF f;
      f.u[0] = *(const uint4*)(qrow + ds * 32 + koff);
      f.u[1] = *(const uint4*)(qrow + ds * 32 + koff + 16);
      qf[ds] = f.v;
    }
  }

  v8f acc[16];
#pragma unroll
  for (int i = 0; i < 16; ++i) acc[i] = v8f_zero();
  float m_run[8], l_run[8];
#pragma unroll
  for (int r = 0; r < 8; ++r) { m_run[r] = -1e30f; l_run[r] = 0.f; }

  const unsigned short* kbase = kc + (size_t)b * S_LEN * DHEAD;
  const unsigned short* vbase = vc + (size_t)b * S_LEN * DHEAD;
  const int t_end = qb * 64 + 64;

  // per-thread K-DMA slice: 128B (8 x b128), 4 threads per key row
  const int ktr  = tid >> 2;
  const int kseg = (tid & 3) * 64;

  for (int t0 = 0; t0 < t_end; t0 += 32) {
    __syncthreads();   // release tile buffers

    // ---- K tile: async global->LDS (no conversion, padded rows) ----
    {
      const unsigned short* gk = kbase + (size_t)(t0 + ktr) * DHEAD + kseg;
      unsigned short* lk = &Klds[ktr][kseg];
#pragma unroll
      for (int j = 0; j < 8; ++j) cp_b128(lk + j * 8, gk + j * 8);
    }
    // ---- V tile: transpose to [d][t] while K DMA is in flight ----
#pragma unroll
    for (int i = 0; i < 64; i += 8) {
      int flat = tid * 64 + i;
      int tr   = flat >> 8;
      int col  = flat & 255;
      uint4 vv = *(const uint4*)(vbase + (size_t)(t0 + tr) * DHEAD + col);
      const unsigned short* pv = (const unsigned short*)&vv;
#pragma unroll
      for (int j = 0; j < 8; ++j) Vlds[col + j][tr] = pv[j];
    }
    WAIT_ASYNC(0);
    __syncthreads();

    // ---- scores: two 16x16 key tiles, contraction over D in 8 steps ----
    v8f sc[2];
    float tmax[8];
#pragma unroll
    for (int r = 0; r < 8; ++r) tmax[r] = -1e30f;
#pragma unroll
    for (int st = 0; st < 2; ++st) {
      v8f s_ = v8f_zero();
#pragma unroll
      for (int ds = 0; ds < 8; ++ds) {
        v16bf kfrag = load_frag(Klds[st * 16 + lcol], ds * 32 + koff);
        s_ = __builtin_amdgcn_wmma_f32_16x16x32_bf16(
            false, qf[ds], false, kfrag, (short)0, s_, false, false);
      }
#pragma unroll
      for (int r = 0; r < 8; ++r) {
        int mg = m0 + r + lhalf8;
        int tg = t0 + st * 16 + lcol;
        if (tg > mg) s_[r] += -1.0e9f;   // reference's additive causal mask
      }
      sc[st] = s_;
#pragma unroll
      for (int r = 0; r < 8; ++r) {      // row max within 16-lane half
        float v = s_[r];
        v = fmaxf(v, __shfl_xor(v, 8, 32));
        v = fmaxf(v, __shfl_xor(v, 4, 32));
        v = fmaxf(v, __shfl_xor(v, 2, 32));
        v = fmaxf(v, __shfl_xor(v, 1, 32));
        tmax[r] = fmaxf(tmax[r], v);
      }
    }

    // ---- online softmax update ----
    float mnew[8], corr[8];
#pragma unroll
    for (int r = 0; r < 8; ++r) {
      mnew[r] = fmaxf(m_run[r], tmax[r]);
      corr[r] = __expf(m_run[r] - mnew[r]);
      m_run[r] = mnew[r];
      l_run[r] *= corr[r];
    }
#pragma unroll
    for (int dt = 0; dt < 16; ++dt)
#pragma unroll
      for (int r = 0; r < 8; ++r) acc[dt][r] *= corr[r];

    // ---- P = exp(sc - m): C-layout -> A-fragment relayout via LDS ----
#pragma unroll
    for (int st = 0; st < 2; ++st) {
#pragma unroll
      for (int r = 0; r < 8; ++r) {
        float p = __expf(sc[st][r] - mnew[r]);
        float s = p;
        s += __shfl_xor(s, 8, 32);
        s += __shfl_xor(s, 4, 32);
        s += __shfl_xor(s, 2, 32);
        s += __shfl_xor(s, 1, 32);
        l_run[r] += s;
        Plds[wv_][r + lhalf8][st * 16 + lcol] = f2bf(p);
      }
    }

    // ---- PV: A = P (16x32 over t), B = V[d][t] tiles ----
    v16bf pf = load_frag(Plds[wv_][lcol], koff);   // in-order LDS within wave
#pragma unroll
    for (int dt = 0; dt < 16; ++dt) {
      v16bf vfrag = load_frag(Vlds[dt * 16 + lcol], koff);
      acc[dt] = __builtin_amdgcn_wmma_f32_16x16x32_bf16(
          false, pf, false, vfrag, (short)0, acc[dt], false, false);
    }
  }

  // ---- normalize + write bf16 [B,S,H*D] (feeds the wo GEMM) ----
  float inv[8];
#pragma unroll
  for (int r = 0; r < 8; ++r) inv[r] = 1.f / l_run[r];
#pragma unroll
  for (int dt = 0; dt < 16; ++dt)
#pragma unroll
    for (int r = 0; r < 8; ++r) {
      int mg = m0 + r + lhalf8;
      out[((size_t)(b * S_LEN + mg)) * (HEADS * DHEAD) + h * DHEAD + dt * 16 + lcol] =
          f2bf(acc[dt][r] * inv[r]);
    }
}

// ---------------- host launch ----------------
extern "C" void kernel_launch(void* const* d_in, const int* in_sizes, int n_in,
                              void* d_out, int out_size, void* d_ws, size_t ws_size,
                              hipStream_t stream) {
  (void)in_sizes; (void)n_in; (void)out_size; (void)ws_size;
  const float* hidden = (const float*)d_in[0];
  const float* wq     = (const float*)d_in[1];
  const float* wk     = (const float*)d_in[2];
  const float* wv     = (const float*)d_in[3];
  const float* wo     = (const float*)d_in[4];
  const float* fcos   = (const float*)d_in[5];
  const float* fsin   = (const float*)d_in[6];
  float* out = (float*)d_out;

  const size_t M = (size_t)BATCH * S_LEN;          // 4096
  const size_t NQ = M * (HEADS * DHEAD);           // 8388608
  const size_t NK = M * DHEAD;                     // 1048576
  const size_t NW = (size_t)HID * HID;             // 4194304
  const size_t NWK = (size_t)DHEAD * HID;          // 524288

  // fp32 region
  float* xq = (float*)d_ws;              // [4096,2048]
  float* xk = xq + NQ;                   // [4096, 256]
  float* xv = xk + NK;                   // [4096, 256]
  // bf16 region
  unsigned short* hb  = (unsigned short*)(xv + NK);   // hidden bf16
  unsigned short* wqb = hb + NQ;
  unsigned short* wkb = wqb + NW;
  unsigned short* wvb = wkb + NWK;
  unsigned short* wob = wvb + NWK;
  unsigned short* qb  = wob + NW;        // scaled Q bf16
  unsigned short* kb  = qb + NQ;
  unsigned short* vb  = kb + NK;
  unsigned short* aob = vb + NK;         // attention out bf16

  dim3 cblk(256);
  cvt_bf16_kernel<<<(int)(NQ / 8 / 256), cblk, 0, stream>>>(hb, hidden, 1.f, (int)(NQ / 8));
  cvt_bf16_kernel<<<(int)(NW / 8 / 256), cblk, 0, stream>>>(wqb, wq, 1.f, (int)(NW / 8));
  cvt_bf16_kernel<<<(int)(NWK / 8 / 256), cblk, 0, stream>>>(wkb, wk, 1.f, (int)(NWK / 8));
  cvt_bf16_kernel<<<(int)(NWK / 8 / 256), cblk, 0, stream>>>(wvb, wv, 1.f, (int)(NWK / 8));
  cvt_bf16_kernel<<<(int)(NW / 8 / 256), cblk, 0, stream>>>(wob, wo, 1.f, (int)(NW / 8));

  gemm_bf16_wmma<<<dim3(HID / TN, M / TM), cblk, 0, stream>>>(xq, hb, wqb, M, HID, HID);
  gemm_bf16_wmma<<<dim3(DHEAD / TN, M / TM), cblk, 0, stream>>>(xk, hb, wkb, M, DHEAD, HID);
  gemm_bf16_wmma<<<dim3(DHEAD / TN, M / TM), cblk, 0, stream>>>(xv, hb, wvb, M, DHEAD, HID);

  int qtot = (int)(M * HEADS * 128);
  rope_kernel<<<(qtot + 255) / 256, 256, 0, stream>>>(xq, fcos, fsin, HEADS, qtot);
  int ktot = (int)(M * 128);
  rope_kernel<<<(ktot + 255) / 256, 256, 0, stream>>>(xk, fcos, fsin, 1, ktot);

  cvt_bf16_kernel<<<(int)(NQ / 8 / 256), cblk, 0, stream>>>(qb, xq, QK_SCALE, (int)(NQ / 8));
  cvt_bf16_kernel<<<(int)(NK / 8 / 256), cblk, 0, stream>>>(kb, xk, 1.f, (int)(NK / 8));
  cvt_bf16_kernel<<<(int)(NK / 8 / 256), cblk, 0, stream>>>(vb, xv, 1.f, (int)(NK / 8));

  attn_kernel<<<dim3(S_LEN / 64, HEADS, BATCH), dim3(128), 0, stream>>>(aob, qb, kb, vb);

  gemm_bf16_wmma<<<dim3(HID / TN, M / TM), cblk, 0, stream>>>(out, aob, wob, M, HID, HID);
}
